// MeshGraphNet_23029614641453
// MI455X (gfx1250) — compile-verified, matrix operand order checked by reading, not run on hardware
//
#include <hip/hip_runtime.h>
#include <hip/hip_bf16.h>

#define HID    128
#define NNODES 10000
#define NEDGES 60000
#define PSTEPS 15

typedef __attribute__((ext_vector_type(16))) __bf16 v16bf;
typedef __attribute__((ext_vector_type(8)))  float  v8f;
typedef __attribute__((ext_vector_type(4)))  unsigned int v4u;

static __device__ __forceinline__ unsigned short f2bfu(float f) {
    union { float f; unsigned u; } c; c.f = f;
    unsigned u = c.u + 0x7FFFu + ((c.u >> 16) & 1u);   // round-to-nearest-even
    return (unsigned short)(u >> 16);
}

union Frag { v16bf v; v4u u4[2]; unsigned short s[16]; };

// Inverse of the ISA 16-bit A/B fragment K layout.
// Forward: elem e (0..15) of a lane holds K = (q<4 ? 2q : 16+2(q-4)) + 8*half + (e&1), q=e>>1.
// Here: kk (0..31) -> (half, elem) so loaders can scatter into fragment order.
static __device__ __forceinline__ void kinv(int kk, int& half, int& elem) {
    int region = kk >> 3;          // 0..3
    int base   = kk & 7;
    half = region & 1;
    int q = ((region >> 1) << 2) + (base >> 1);
    elem = 2 * q + (base & 1);
}

// Generic fused MeshGraphMLP block:
//   y = LayerNorm(relu(concat(A0, A1[idx1], A2[idx2]) @ W0 + b0) @ W1 + b1) * gam + bet [+ resid]
// 128-row tile per block, 256 threads (8 waves), bf16 WMMA f32-accumulate.
// K1 (input dim) and N2 (output dim) are compile-time so all tile loops unroll
// and accumulators stay in VGPRs (no scratch spills).
template<int K1, int N2>
__global__ __launch_bounds__(256)
void mlp_block(const float* __restrict__ A0, int d0,
               const float* __restrict__ A1, const int* __restrict__ idx1, int d1,
               const float* __restrict__ A2, const int* __restrict__ idx2, int d2,
               const float* __restrict__ W0, const float* __restrict__ b0,
               const float* __restrict__ W1, const float* __restrict__ b1,
               const float* __restrict__ gam, const float* __restrict__ bet,
               const float* __restrict__ resid, float* __restrict__ out, int M)
{
    constexpr int NCH1 = (K1 + 31) / 32;   // GEMM1 K chunks
    constexpr int NT2  = (N2 + 15) / 16;   // GEMM2 output col tiles

    // All LDS tiles are stored in WMMA *fragment order*: [tile][lane][elem16],
    // so a fragment load is two contiguous ds_load_b128 per lane.
    __shared__ union {
        struct {
            unsigned short AldF[8][32][16];    // A chunk frags (wave-indexed), 8 KB
            unsigned short BldF[8][32][16];    // W chunk frags (col-tile-indexed), 8 KB
            unsigned short hF[4][8][32][16];   // hidden acts as GEMM2 A-frags, 32 KB
        } s;
        float ylds[128][128];                  // pre-LN output (after GEMM2), 64 KB
    } L;

    const int tid  = threadIdx.x;
    const int lane = tid & 31;
    const int wv   = tid >> 5;      // wave 0..7
    const int half = lane >> 4;
    const int base = blockIdx.x * 128;

    v8f acc[8];
#pragma unroll
    for (int t = 0; t < 8; ++t)
#pragma unroll
        for (int i = 0; i < 8; ++i) acc[t][i] = 0.f;

    // ---------------- GEMM1: A(128 x K1) @ W0(K1 x 128), K chunked by 32 ----------------
    for (int kc = 0; kc < NCH1; ++kc) {
        // cooperative gather + concat + cvt of A chunk, scattered into frag order
#pragma unroll
        for (int i = 0; i < 16; ++i) {
            int idx = tid + i * 256;
            int r = idx >> 5, kk = idx & 31;
            int g  = base + r;
            int gk = kc * 32 + kk;
            float v = 0.f;
            if (g < M && gk < K1) {
                if (gk < d0)            v = A0[(long)g * d0 + gk];
                else if (gk < d0 + d1) {
                    int row = idx1 ? idx1[g] : g;
                    v = A1[(long)row * d1 + (gk - d0)];
                } else {
                    int row = idx2 ? idx2[g] : g;
                    v = A2[(long)row * d2 + (gk - d0 - d1)];
                }
            }
            int hf, el; kinv(kk, hf, el);
            L.s.AldF[r >> 4][hf * 16 + (r & 15)][el] = f2bfu(v);
        }
        // cooperative W0 chunk (zero-padded past K1), scattered into frag order
#pragma unroll
        for (int i = 0; i < 16; ++i) {
            int idx = tid + i * 256;
            int kk = idx >> 7, n = idx & 127;
            int gk = kc * 32 + kk;
            float v = (gk < K1) ? W0[(long)gk * HID + n] : 0.f;
            int hf, el; kinv(kk, hf, el);
            L.s.BldF[n >> 4][hf * 16 + (n & 15)][el] = f2bfu(v);
        }
        __syncthreads();
        Frag af;
        {
            const v4u* p = (const v4u*)L.s.AldF[wv][lane];
            af.u4[0] = p[0]; af.u4[1] = p[1];        // 2x ds_load_b128
        }
#pragma unroll
        for (int t = 0; t < 8; ++t) {
            Frag bf;
            const v4u* p = (const v4u*)L.s.BldF[t][lane];
            bf.u4[0] = p[0]; bf.u4[1] = p[1];
            acc[t] = __builtin_amdgcn_wmma_f32_16x16x32_bf16(
                false, af.v, false, bf.v, (short)0, acc[t], false, false);
        }
        __syncthreads();
    }

    // ---------------- bias + ReLU -> hidden, stored directly as GEMM2 A-frags ----------------
    const int lm = lane & 15;
#pragma unroll
    for (int t = 0; t < 8; ++t) {
        int col = t * 16 + lm;
        float bb = b0[col];
        int c2 = col >> 5, kk = col & 31;
        int hf, el; kinv(kk, hf, el);
#pragma unroll
        for (int j = 0; j < 8; ++j) {
            int rl = j + 8 * half;                   // row within wave's 16 rows (C/D layout)
            float hv = acc[t][j] + bb;
            hv = hv > 0.f ? hv : 0.f;
            L.s.hF[c2][wv][hf * 16 + rl][el] = f2bfu(hv);
        }
    }

    // ---------------- GEMM2: h(128 x 128) @ W1(128 x N2) ----------------
#pragma unroll
    for (int t = 0; t < 8; ++t)
#pragma unroll
        for (int i = 0; i < 8; ++i) acc[t][i] = 0.f;

    for (int c2 = 0; c2 < 4; ++c2) {
#pragma unroll
        for (int i = 0; i < 16; ++i) {
            int idx = tid + i * 256;
            int kk = idx >> 7, n = idx & 127;
            int gk = c2 * 32 + kk;
            float v = (n < N2) ? W1[(long)gk * N2 + n] : 0.f;
            int hf, el; kinv(kk, hf, el);
            L.s.BldF[n >> 4][hf * 16 + (n & 15)][el] = f2bfu(v);
        }
        __syncthreads();
        Frag af;
        {
            const v4u* p = (const v4u*)L.s.hF[c2][wv][lane];
            af.u4[0] = p[0]; af.u4[1] = p[1];
        }
#pragma unroll
        for (int t = 0; t < NT2; ++t) {
            Frag bf;
            const v4u* p = (const v4u*)L.s.BldF[t][lane];
            bf.u4[0] = p[0]; bf.u4[1] = p[1];
            acc[t] = __builtin_amdgcn_wmma_f32_16x16x32_bf16(
                false, af.v, false, bf.v, (short)0, acc[t], false, false);
        }
        __syncthreads();
    }

    // ---------------- y (fp32) to LDS (aliases bf16 staging; all reads done) ----------------
#pragma unroll
    for (int t = 0; t < NT2; ++t) {
        int col = t * 16 + lm;
        float bb = (col < N2) ? b1[col] : 0.f;
#pragma unroll
        for (int j = 0; j < 8; ++j) {
            int row = wv * 16 + j + 8 * half;
            L.ylds[row][col] = acc[t][j] + bb;
        }
    }
    __syncthreads();

    // ---------------- LayerNorm + residual + store, one row per thread ----------------
    if (tid < 128) {
        int g = base + tid;
        if (g < M) {
            float m = 0.f;
            for (int c = 0; c < N2; ++c) m += L.ylds[tid][c];
            m /= (float)N2;
            float var = 0.f;
            for (int c = 0; c < N2; ++c) { float d = L.ylds[tid][c] - m; var += d * d; }
            var /= (float)N2;
            float rs = rsqrtf(var + 1e-5f);
            for (int c = 0; c < N2; ++c) {
                float o = (L.ylds[tid][c] - m) * rs * gam[c] + bet[c];
                if (resid) o += resid[(long)g * N2 + c];
                out[(long)g * N2 + c] = o;
            }
        }
    }
}

// agg[dst[e]] += e_feat[e]  (fp32 global atomics; one element per thread)
__global__ __launch_bounds__(256)
void scatter_add(const float* __restrict__ e, const int* __restrict__ dst,
                 float* __restrict__ agg, int E)
{
    int i = blockIdx.x * 256 + threadIdx.x;
    if (i < E * HID) {
        int eid = i >> 7;
        int j   = i & 127;
        unsafeAtomicAdd(&agg[(long)dst[eid] * HID + j], e[i]);
    }
}

extern "C" void kernel_launch(void* const* d_in, const int* in_sizes, int n_in,
                              void* d_out, int out_size, void* d_ws, size_t ws_size,
                              hipStream_t stream)
{
    const float* nodef = (const float*)d_in[0];
    const float* edgef = (const float*)d_in[1];
    const int*   src   = (const int*)d_in[2];
    const int*   dst   = (const int*)d_in[3];

    // params pytree, sorted keys: dec, enc_e, enc_n, proc_e, proc_n; each {b0,b1,bt,g,w0,w1}
    const float* dec_b0 = (const float*)d_in[4];
    const float* dec_b1 = (const float*)d_in[5];
    const float* dec_bt = (const float*)d_in[6];
    const float* dec_g  = (const float*)d_in[7];
    const float* dec_w0 = (const float*)d_in[8];
    const float* dec_w1 = (const float*)d_in[9];
    const float* ee_b0  = (const float*)d_in[10];
    const float* ee_b1  = (const float*)d_in[11];
    const float* ee_bt  = (const float*)d_in[12];
    const float* ee_g   = (const float*)d_in[13];
    const float* ee_w0  = (const float*)d_in[14];
    const float* ee_w1  = (const float*)d_in[15];
    const float* en_b0  = (const float*)d_in[16];
    const float* en_b1  = (const float*)d_in[17];
    const float* en_bt  = (const float*)d_in[18];
    const float* en_g   = (const float*)d_in[19];
    const float* en_w0  = (const float*)d_in[20];
    const float* en_w1  = (const float*)d_in[21];
    const float* pe_b0  = (const float*)d_in[22];
    const float* pe_b1  = (const float*)d_in[23];
    const float* pe_bt  = (const float*)d_in[24];
    const float* pe_g   = (const float*)d_in[25];
    const float* pe_w0  = (const float*)d_in[26];
    const float* pe_w1  = (const float*)d_in[27];
    const float* pn_b0  = (const float*)d_in[28];
    const float* pn_b1  = (const float*)d_in[29];
    const float* pn_bt  = (const float*)d_in[30];
    const float* pn_g   = (const float*)d_in[31];
    const float* pn_w0  = (const float*)d_in[32];
    const float* pn_w1  = (const float*)d_in[33];

    float* x   = (float*)d_ws;                 // [NNODES, HID]
    float* e   = x + (size_t)NNODES * HID;     // [NEDGES, HID]
    float* agg = e + (size_t)NEDGES * HID;     // [NNODES, HID]

    dim3 blk(256);
    const int gN = (NNODES + 127) / 128;
    const int gE = (NEDGES + 127) / 128;
    const int gS = (NEDGES * HID + 255) / 256;

    // encoders
    mlp_block<12, 128><<<gN, blk, 0, stream>>>(nodef, 12, nullptr, nullptr, 0,
        nullptr, nullptr, 0,
        en_w0, en_b0, en_w1, en_b1, en_g, en_bt, nullptr, x, NNODES);
    mlp_block<4, 128><<<gE, blk, 0, stream>>>(edgef, 4, nullptr, nullptr, 0,
        nullptr, nullptr, 0,
        ee_w0, ee_b0, ee_w1, ee_b1, ee_g, ee_bt, nullptr, e, NEDGES);

    // 15 processor steps
    for (int p = 0; p < PSTEPS; ++p) {
        // edge block: e = MLP(e || x[src] || x[dst]) + e    (in-place, row-partitioned)
        mlp_block<384, 128><<<gE, blk, 0, stream>>>(e, HID, x, src, HID, x, dst, HID,
            pe_w0 + (size_t)p * 384 * HID, pe_b0 + (size_t)p * HID,
            pe_w1 + (size_t)p * HID * HID, pe_b1 + (size_t)p * HID,
            pe_g + (size_t)p * HID, pe_bt + (size_t)p * HID,
            e, e, NEDGES);
        // segment_sum(e -> dst)
        hipMemsetAsync(agg, 0, (size_t)NNODES * HID * sizeof(float), stream);
        scatter_add<<<gS, blk, 0, stream>>>(e, dst, agg, NEDGES);
        // node block: x = MLP(agg || x) + x                 (in-place, row-partitioned)
        mlp_block<256, 128><<<gN, blk, 0, stream>>>(agg, HID, x, nullptr, HID,
            nullptr, nullptr, 0,
            pn_w0 + (size_t)p * 256 * HID, pn_b0 + (size_t)p * HID,
            pn_w1 + (size_t)p * HID * HID, pn_b1 + (size_t)p * HID,
            pn_g + (size_t)p * HID, pn_bt + (size_t)p * HID,
            x, x, NNODES);
    }

    // decoder -> d_out [NNODES, 3]
    mlp_block<128, 3><<<gN, blk, 0, stream>>>(x, HID, nullptr, nullptr, 0,
        nullptr, nullptr, 0,
        dec_w0, dec_b0, dec_w1, dec_b1, dec_g, dec_bt, nullptr, (float*)d_out, NNODES);
}
